// NodeClassifier_8297876816436
// MI455X (gfx1250) — compile-verified, matrix operand order checked by reading, not run on hardware
//
#include <hip/hip_runtime.h>

// ---------------- types for WMMA ----------------
typedef __attribute__((ext_vector_type(16))) __bf16 v16bf;
typedef __attribute__((ext_vector_type(8)))  float  v8f;

// ---------------- small device helpers ----------------
__device__ __forceinline__ unsigned short f2bf(float f) {
    unsigned u = __float_as_uint(f);
    u += 0x7fffu + ((u >> 16) & 1u);          // round-to-nearest-even
    return (unsigned short)(u >> 16);
}
__device__ __forceinline__ float bf2f(unsigned short h) {
    return __uint_as_float(((unsigned)h) << 16);
}
// order-preserving float <-> uint encoding for atomicMax on floats
__device__ __forceinline__ unsigned fenc(float f) {
    unsigned u = __float_as_uint(f);
    return (u & 0x80000000u) ? ~u : (u | 0x80000000u);
}
__device__ __forceinline__ float fdec(unsigned k) {
    unsigned u = (k & 0x80000000u) ? (k & 0x7fffffffu) : ~k;
    return __uint_as_float(u);
}

// ---------------- elementwise / prep kernels ----------------
__global__ void k_f32_to_bf16(const float* __restrict__ in, unsigned short* __restrict__ out, int n) {
    int i = blockIdx.x * blockDim.x + threadIdx.x;
    if (i < n) out[i] = f2bf(in[i]);
}

__global__ void k_zero_f32(float* __restrict__ p, int n) {
    int i = blockIdx.x * blockDim.x + threadIdx.x;
    if (i < n) p[i] = 0.0f;
}

__global__ void k_init_softmax(unsigned* __restrict__ umax, float* __restrict__ denom, int n) {
    int i = blockIdx.x * blockDim.x + threadIdx.x;
    if (i < n) { umax[i] = 0x007fffffu /* enc(-inf) */; denom[i] = 0.0f; }
}

// W [cin,cout] f32 row-major -> Wt [cout,cin] bf16 (so B fragments load contiguously)
__global__ void k_transpose_w(const float* __restrict__ W, unsigned short* __restrict__ Wt,
                              int cin, int cout) {
    int i = blockIdx.x * blockDim.x + threadIdx.x;
    if (i < cin * cout) {
        int k = i / cout, n = i % cout;
        Wt[(size_t)n * cin + k] = f2bf(W[i]);
    }
}

// wd_ad[k] = sum_j Wd[k,j] * ad[j]
__global__ void k_wd_ad(const float* __restrict__ Wd, const float* __restrict__ ad,
                        float* __restrict__ wd_ad, int cin, int cout) {
    int k = blockIdx.x * blockDim.x + threadIdx.x;
    if (k < cin) {
        float s = 0.0f;
        const float* row = Wd + (size_t)k * cout;
        for (int j = 0; j < cout; ++j) s += row[j] * ad[j];
        wd_ad[k] = s;
    }
}

// ---------------- WMMA bf16 GEMM: C[M,Ncols] = A[M,K] * Bt[Ncols,K]^T ----------------
// One wave computes a 16x64 output strip (4 x 16x16 tiles, A-fragment reused).
// Requires M%16==0, K%32==0, Ncols%64==0 (true for all dims here).
__global__ void k_wmma_gemm(const unsigned short* __restrict__ A,
                            const unsigned short* __restrict__ Bt,
                            float* __restrict__ C,
                            int M, int Ncols, int K) {
    const int lane = threadIdx.x & 31;
    const int wave = threadIdx.x >> 5;
    const int wtiles_n = Ncols >> 6;
    const int total = (M >> 4) * wtiles_n;
    const int wt = blockIdx.x * (blockDim.x >> 5) + wave;
    if (wt >= total) return;                      // wave-uniform: EXEC stays all-ones
    const int tm = wt / wtiles_n;
    const int tn = wt % wtiles_n;

    const int l15 = lane & 15;
    const int khi = (lane >> 4) << 3;             // lanes 16-31 hold K+8 halves
    const size_t arow = (size_t)(tm * 16 + l15) * K;
    size_t brow[4];
#pragma unroll
    for (int s = 0; s < 4; ++s)
        brow[s] = (size_t)(tn * 64 + s * 16 + l15) * K;

    v8f acc[4] = {v8f{}, v8f{}, v8f{}, v8f{}};

    for (int kb = 0; kb < K; kb += 32) {
        union { v16bf v; unsigned u[8]; } af;
#pragma unroll
        for (int q = 0; q < 8; ++q) {
            int k0 = kb + ((q & 3) << 1) + ((q >> 2) << 4) + khi;   // ISA 16-bit A layout
            af.u[q] = *(const unsigned*)(A + arow + k0);
        }
#pragma unroll
        for (int s = 0; s < 4; ++s) {
            union { v16bf v; unsigned u[8]; } bf;
#pragma unroll
            for (int q = 0; q < 8; ++q) {
                int k0 = kb + ((q & 3) << 1) + ((q >> 2) << 4) + khi;
                bf.u[q] = *(const unsigned*)(Bt + brow[s] + k0);
            }
            acc[s] = __builtin_amdgcn_wmma_f32_16x16x32_bf16(
                false, af.v, false, bf.v, (short)0, acc[s], false, false);
        }
    }
    // C/D layout: VGPR q -> M = q + 8*(lane>=16), N = lane&15
    const int mhi = (lane >> 4) << 3;
#pragma unroll
    for (int s = 0; s < 4; ++s) {
        int nb = tn * 64 + s * 16 + l15;
#pragma unroll
        for (int q = 0; q < 8; ++q) {
            int mm = tm * 16 + q + mhi;
            C[(size_t)mm * Ncols + nb] = acc[s][q];
        }
    }
}

// ---------------- wave-per-row dot products ----------------
__global__ void k_rowdot_f32(const float* __restrict__ X, const float* __restrict__ w,
                             float* __restrict__ out, int nrows, int cols) {
    int wid = blockIdx.x * (blockDim.x >> 5) + (threadIdx.x >> 5);
    int lane = threadIdx.x & 31;
    if (wid >= nrows) return;
    const float* row = X + (size_t)wid * cols;
    float s = 0.0f;
    for (int c = lane; c < cols; c += 32) s += row[c] * w[c];
    for (int o = 16; o; o >>= 1) s += __shfl_down(s, o, 32);
    if (lane == 0) out[wid] = s;
}

__global__ void k_rowdot_bf16(const unsigned short* __restrict__ X, const float* __restrict__ w,
                              float* __restrict__ out, int nrows, int cols) {
    int wid = blockIdx.x * (blockDim.x >> 5) + (threadIdx.x >> 5);
    int lane = threadIdx.x & 31;
    if (wid >= nrows) return;
    const unsigned short* row = X + (size_t)wid * cols;
    float s = 0.0f;
    for (int c = lane; c < cols; c += 32) s += bf2f(row[c]) * w[c];
    for (int o = 16; o; o >>= 1) s += __shfl_down(s, o, 32);
    if (lane == 0) out[wid] = s;
}

// ---------------- edge-stage kernels ----------------
__global__ void k_edge_logits(const int* __restrict__ src, const int* __restrict__ dst,
                              const float* __restrict__ a_src, const float* __restrict__ a_dst,
                              float* __restrict__ logits, unsigned* __restrict__ umax, int E_) {
    int e = blockIdx.x * blockDim.x + threadIdx.x;
    if (e < E_) {
        float l = a_src[src[e]] + a_dst[dst[e]];
        l = (l > 0.0f) ? l : 0.2f * l;            // LeakyReLU(0.2)
        logits[e] = l;
        atomicMax(&umax[dst[e]], fenc(l));
    }
}

__global__ void k_edge_exp(const int* __restrict__ dst, float* __restrict__ logits,
                           const unsigned* __restrict__ umax, float* __restrict__ denom, int E_) {
    int e = blockIdx.x * blockDim.x + threadIdx.x;
    if (e < E_) {
        int d = dst[e];
        float v = __expf(logits[e] - fdec(umax[d]));
        logits[e] = v;                            // reuse buffer for exp values
        atomicAdd(&denom[d], v);
    }
}

// wave per edge: acc[dst,:] += alpha * hs[src,:]
__global__ void k_edge_scatter(const int* __restrict__ src, const int* __restrict__ dst,
                               const float* __restrict__ eval, const float* __restrict__ denom,
                               const float* __restrict__ hs, float* __restrict__ acc,
                               int E_, int cout) {
    int wid = blockIdx.x * (blockDim.x >> 5) + (threadIdx.x >> 5);
    int lane = threadIdx.x & 31;
    if (wid >= E_) return;
    int s = src[wid], d = dst[wid];
    float dn = denom[d];
    dn = (dn > 1e-16f) ? dn : 1e-16f;
    float alpha = eval[wid] / dn;
    const float* hrow = hs + (size_t)s * cout;
    float* arow = acc + (size_t)d * cout;
    for (int c = lane; c < cout; c += 32)
        atomicAdd(&arow[c], alpha * hrow[c]);
}

// out_bf16 = relu(scale * (acc + b0 + b1)); handles HeteroConv mean via scale=0.5
__global__ void k_epilogue(const float* __restrict__ acc, const float* __restrict__ b0,
                           const float* __restrict__ b1, float scale,
                           unsigned short* __restrict__ xout, int nrows, int cout) {
    int i = blockIdx.x * blockDim.x + threadIdx.x;
    if (i < nrows * cout) {
        int c = i % cout;
        float b = b0[c] + (b1 ? b1[c] : 0.0f);
        float v = scale * (acc[i] + b);
        xout[i] = f2bf(v > 0.0f ? v : 0.0f);
    }
}

// final lin2: [N,256] bf16 @ [256,2] f32 + b2 -> d_out f32 [N,2], wave per row
__global__ void k_head(const unsigned short* __restrict__ H, const float* __restrict__ W2,
                       const float* __restrict__ b2, float* __restrict__ out,
                       int nrows, int cols) {
    int wid = blockIdx.x * (blockDim.x >> 5) + (threadIdx.x >> 5);
    int lane = threadIdx.x & 31;
    if (wid >= nrows) return;
    const unsigned short* row = H + (size_t)wid * cols;
    float s0 = 0.0f, s1 = 0.0f;
    for (int c = lane; c < cols; c += 32) {
        float h = bf2f(row[c]);
        s0 += h * W2[c * 2 + 0];
        s1 += h * W2[c * 2 + 1];
    }
    for (int o = 16; o; o >>= 1) { s0 += __shfl_down(s0, o, 32); s1 += __shfl_down(s1, o, 32); }
    if (lane == 0) { out[wid * 2 + 0] = s0 + b2[0]; out[wid * 2 + 1] = s1 + b2[1]; }
}

// ---------------- host orchestration ----------------
static inline int cdiv(int a, int b) { return (a + b - 1) / b; }

extern "C" void kernel_launch(void* const* d_in, const int* in_sizes, int n_in,
                              void* d_out, int out_size, void* d_ws, size_t ws_size,
                              hipStream_t stream) {
    const int DIMS[6] = {64, 128, 256, 512, 256, 128};
    const int N  = in_sizes[0] / DIMS[0];     // 50000
    const int E  = in_sizes[3] / 2;           // 150000
    const int DMAX = 512;

    const float* x_in[3] = {(const float*)d_in[0], (const float*)d_in[1], (const float*)d_in[2]};
    const int*   ei[4]   = {(const int*)d_in[3], (const int*)d_in[4],
                            (const int*)d_in[5], (const int*)d_in[6]};
    // params flattened in dict order: 5 layers x 4 rels(ne,en,sn,ns) x 5 (Ws,Wd,as,ad,b)
    auto P = [&](int layer, int rel, int t) -> const float* {
        return (const float*)d_in[7 + layer * 20 + rel * 5 + t];
    };
    const float* lin1_w = (const float*)d_in[107];
    const float* lin1_b = (const float*)d_in[108];
    const float* lin2_w = (const float*)d_in[109];
    const float* lin2_b = (const float*)d_in[110];

    // ---- workspace carve ----
    char*  ws  = (char*)d_ws;
    size_t off = 0;
    auto carve = [&](size_t bytes) -> char* {
        char* p = ws + off;
        off = (off + bytes + 255) & ~(size_t)255;
        return p;
    };
    unsigned short* xbuf[2][3];               // [ping-pong][node,edge,source], bf16 activations
    for (int b = 0; b < 2; ++b)
        for (int t = 0; t < 3; ++t)
            xbuf[b][t] = (unsigned short*)carve((size_t)N * DMAX * 2);
    float* acc[3];
    for (int t = 0; t < 3; ++t) acc[t] = (float*)carve((size_t)N * DMAX * 4);
    float*          hs     = (float*)carve((size_t)N * DMAX * 4);
    unsigned short* Wt     = (unsigned short*)carve((size_t)DMAX * DMAX * 2);
    float*          wd_ad  = (float*)carve((size_t)DMAX * 4);
    float*          a_src  = (float*)carve((size_t)N * 4);
    float*          a_dst  = (float*)carve((size_t)N * 4);
    float*          logits = (float*)carve((size_t)E * 4);
    unsigned*       umax   = (unsigned*)carve((size_t)N * 4);
    float*          denom  = (float*)carve((size_t)N * 4);
    (void)ws_size; (void)n_in; (void)out_size;

    // ---- convert layer-0 inputs to bf16 ----
    for (int t = 0; t < 3; ++t) {
        int n = N * DIMS[0];
        k_f32_to_bf16<<<cdiv(n, 256), 256, 0, stream>>>(x_in[t], xbuf[0][t], n);
    }

    // relation tables: src type, dst type, edge-index id, accumulator target, zero-first?
    const int r_src[4] = {0, 1, 2, 0};
    const int r_dst[4] = {1, 0, 0, 2};
    const int r_acc[4] = {1, 0, 0, 2};
    const int r_zero[4] = {1, 1, 0, 1};

    for (int L = 0; L < 5; ++L) {
        const int cin = DIMS[L], cout = DIMS[L + 1];
        const int cur = L & 1, nxt = cur ^ 1;

        for (int r = 0; r < 4; ++r) {
            const float* Wsrc = P(L, r, 0);
            const float* Wdst = P(L, r, 1);
            const float* att_s = P(L, r, 2);
            const float* att_d = P(L, r, 3);
            const int* e_src = ei[r];
            const int* e_dst = ei[r] + E;
            float* accT = acc[r_acc[r]];

            // weight prep
            k_transpose_w<<<cdiv(cin * cout, 256), 256, 0, stream>>>(Wsrc, Wt, cin, cout);
            k_wd_ad<<<cdiv(cin, 64), 64, 0, stream>>>(Wdst, att_d, wd_ad, cin, cout);

            // hs = x_src @ Ws  (bf16 WMMA, f32 accum)
            int wtiles = (N / 16) * (cout / 64);
            k_wmma_gemm<<<cdiv(wtiles, 8), 256, 0, stream>>>(
                xbuf[cur][r_src[r]], Wt, hs, N, cout, cin);

            // attention scalar terms
            k_rowdot_f32<<<cdiv(N, 8), 256, 0, stream>>>(hs, att_s, a_src, N, cout);
            k_rowdot_bf16<<<cdiv(N, 8), 256, 0, stream>>>(xbuf[cur][r_dst[r]], wd_ad, a_dst, N, cin);

            // segment softmax + scatter
            k_init_softmax<<<cdiv(N, 256), 256, 0, stream>>>(umax, denom, N);
            if (r_zero[r])
                k_zero_f32<<<cdiv(N * cout, 256), 256, 0, stream>>>(accT, N * cout);
            k_edge_logits<<<cdiv(E, 256), 256, 0, stream>>>(e_src, e_dst, a_src, a_dst,
                                                            logits, umax, E);
            k_edge_exp<<<cdiv(E, 256), 256, 0, stream>>>(e_dst, logits, umax, denom, E);
            k_edge_scatter<<<cdiv(E, 8), 256, 0, stream>>>(e_src, e_dst, logits, denom,
                                                           hs, accT, E, cout);
        }

        // epilogues: bias (+mean for 'node'), ReLU, requantize to bf16
        int n = N * cout;
        k_epilogue<<<cdiv(n, 256), 256, 0, stream>>>(acc[1], P(L, 0, 4), nullptr, 1.0f,
                                                     xbuf[nxt][1], N, cout);   // edge  <- ne
        k_epilogue<<<cdiv(n, 256), 256, 0, stream>>>(acc[0], P(L, 1, 4), P(L, 2, 4), 0.5f,
                                                     xbuf[nxt][0], N, cout);   // node  <- mean(en,sn)
        k_epilogue<<<cdiv(n, 256), 256, 0, stream>>>(acc[2], P(L, 3, 4), nullptr, 1.0f,
                                                     xbuf[nxt][2], N, cout);   // source<- ns
    }

    // ---- head: h = relu(x_node @ lin1_w + lin1_b); out = h @ lin2_w + lin2_b ----
    unsigned short* x_node_final = xbuf[1][0];   // after 5 layers (L=4: nxt=1), dim 128
    k_transpose_w<<<cdiv(128 * 256, 256), 256, 0, stream>>>(lin1_w, Wt, 128, 256);
    int wtiles = (N / 16) * (256 / 64);
    k_wmma_gemm<<<cdiv(wtiles, 8), 256, 0, stream>>>(x_node_final, Wt, acc[0], N, 256, 128);
    k_epilogue<<<cdiv(N * 256, 256), 256, 0, stream>>>(acc[0], lin1_b, nullptr, 1.0f,
                                                       xbuf[0][0], N, 256);
    k_head<<<cdiv(N, 8), 256, 0, stream>>>(xbuf[0][0], lin2_w, lin2_b, (float*)d_out, N, 256);
}
